// GCN_51891794870618
// MI455X (gfx1250) — matrix operand order repaired and checked by reading, not compiled, not run on hardware
//
#include <hip/hip_runtime.h>
#include <hip/hip_bf16.h>
#include <math.h>

// ---------------- problem dimensions (compile-time) ----------------
#define N_NODES 8192
#define E_EDGES 262144
#define IN_DIM  3000
#define IN_PAD  3008          // 94 * 32, zero-padded K for WMMA
#define H1D 512
#define H2D 256
#define H3D 128
#define ZD  64

typedef __attribute__((ext_vector_type(16))) __bf16 v16bf;
typedef __attribute__((ext_vector_type(8)))  __bf16 v8bf;
typedef __attribute__((ext_vector_type(8)))  float  v8f;

union ABFrag { v16bf v; v8bf h[2]; };

// ---------------- WMMA fragment loaders (wave32, contiguous k order, same in A and B) -------
// A (16x32 bf16, row-major [M][K], lda = K stride):
//   lane<16 : elems i = K[k0 + i]      (row m = lane)
//   lane>=16: elems i = K[k0 + 16 + i] (row m = lane-16)
__device__ __forceinline__ v16bf load_a_frag(const __bf16* __restrict__ A,
                                             int lda, int m, int k0, int ksel) {
  ABFrag f;
  const __bf16* p = A + (size_t)m * lda + k0 + 16 * ksel;
  f.h[0] = *(const v8bf*)(p);
  f.h[1] = *(const v8bf*)(p + 8);
  return f.v;
}

// B (32x16 bf16) supplied as Bt[N][K] row-major (i.e. B transposed):
//   lane<16 : elems i = K[k0 + i]      (column n = lane)
//   lane>=16: elems i = K[k0 + 16 + i] (column n = lane-16)
__device__ __forceinline__ v16bf load_b_frag(const __bf16* __restrict__ Bt,
                                             int ldb, int n, int k0, int ksel) {
  const __bf16* p = Bt + (size_t)n * ldb + k0 + 16 * ksel;
  return *(const v16bf*)(p);
}

// ---------------- elementwise / conversion kernels ----------------
__global__ void conv_x_bf16(const float* __restrict__ x, __bf16* __restrict__ xb,
                            long long total) {
  long long t = (long long)blockIdx.x * blockDim.x + threadIdx.x;
  if (t >= total) return;
  int n = (int)(t / IN_PAD);
  int k = (int)(t % IN_PAD);
  float v = (k < IN_DIM) ? x[(size_t)n * IN_DIM + k] : 0.0f;
  xb[t] = (__bf16)v;
}

// W [fi][fo] f32  ->  Wt [fo][fi_pad] bf16 (transposed, K zero-padded)
__global__ void conv_w_bf16_t(const float* __restrict__ W, __bf16* __restrict__ Wt,
                              int fi, int fo, int fi_pad, long long total) {
  long long t = (long long)blockIdx.x * blockDim.x + threadIdx.x;
  if (t >= total) return;
  int o = (int)(t / fi_pad);
  int k = (int)(t % fi_pad);
  float v = (k < fi) ? W[(size_t)k * fo + o] : 0.0f;
  Wt[t] = (__bf16)v;
}

__global__ void bias_init(float* __restrict__ agg, const float* __restrict__ b,
                          int F, long long total) {
  long long t = (long long)blockIdx.x * blockDim.x + threadIdx.x;
  if (t >= total) return;
  agg[t] = b[(int)(t % F)];
}

__global__ void relu_to_bf16(const float* __restrict__ in, __bf16* __restrict__ out,
                             long long total) {
  long long t = (long long)blockIdx.x * blockDim.x + threadIdx.x;
  if (t >= total) return;
  out[t] = (__bf16)fmaxf(in[t], 0.0f);
}

// final layer: write f32 emb to d_out and bf16 copy for the decoder GEMM
__global__ void relu_emb(const float* __restrict__ in, float* __restrict__ emb,
                         __bf16* __restrict__ embb, long long total) {
  long long t = (long long)blockIdx.x * blockDim.x + threadIdx.x;
  if (t >= total) return;
  float v = fmaxf(in[t], 0.0f);
  emb[t] = v;
  embb[t] = (__bf16)v;
}

// ---------------- SpMM: agg[row] += w * support[col]  (atomic scatter) -----
__global__ void spmm_scatter(const int* __restrict__ row, const int* __restrict__ col,
                             const float* __restrict__ ew, const float* __restrict__ S,
                             float* __restrict__ agg, int F, int fshift4,
                             long long total) {
  long long t = (long long)blockIdx.x * blockDim.x + threadIdx.x;
  if (t >= total) return;
  int e = (int)(t >> fshift4);                  // edge index
  int f = (int)(t & ((1 << fshift4) - 1)) << 2; // feature group of 4
  float w = ew[e];
  int c = col[e], r = row[e];
  const float4 s = *(const float4*)(S + (size_t)c * F + f);
  float* d = agg + (size_t)r * F + f;
  atomicAdd(d + 0, w * s.x);
  atomicAdd(d + 1, w * s.y);
  atomicAdd(d + 2, w * s.z);
  atomicAdd(d + 3, w * s.w);
}

// ---------------- WMMA GEMM: C[M][Nf] f32 = A[M][K] bf16 x Bt[Nf][K] bf16 ---
// Register-blocked: each wave computes a 64(M) x 32(N) macro-tile:
// 4 A-fragments x 2 B-fragments -> 8 independent v_wmma per K-step.
// Bytes per WMMA drop from 64B (naive) to 24B; 8 independent D accumulators
// let the XDL pipe run back-to-back without hazard stalls.
__global__ __launch_bounds__(256) void gemm_bf16_wmma(
    const __bf16* __restrict__ A, int lda,
    const __bf16* __restrict__ Bt, int ldb,
    float* __restrict__ C, int ldc,
    int tilesN, int totalTiles, int K) {
  int wave = (int)((blockIdx.x * blockDim.x + threadIdx.x) >> 5);
  if (wave >= totalTiles) return;
  int lane = threadIdx.x & 31;
  int tm = wave / tilesN, tn = wave % tilesN;   // macro-tile coords
  int l15 = lane & 15, ksel = lane >> 4;
  int m0 = tm * 64 + l15;
  int n0 = tn * 32 + l15;

  v8f acc[4][2] = {};
  for (int k0 = 0; k0 < K; k0 += 32) {
    v16bf a[4], b[2];
#pragma unroll
    for (int i = 0; i < 4; ++i)
      a[i] = load_a_frag(A, lda, m0 + 16 * i, k0, ksel);
#pragma unroll
    for (int j = 0; j < 2; ++j)
      b[j] = load_b_frag(Bt, ldb, n0 + 16 * j, k0, ksel);
    __builtin_prefetch(A + (size_t)m0 * lda + k0 + 32, 0, 1);
#pragma unroll
    for (int i = 0; i < 4; ++i)
#pragma unroll
      for (int j = 0; j < 2; ++j)
        acc[i][j] = __builtin_amdgcn_wmma_f32_16x16x32_bf16(
            false, a[i], false, b[j], (short)0, acc[i][j], false, false);
  }
  // C/D layout: VGPR v -> row M = 16*i + 8*ksel + v, col n
#pragma unroll
  for (int i = 0; i < 4; ++i) {
    int mbase = tm * 64 + 16 * i + 8 * ksel;
#pragma unroll
    for (int j = 0; j < 2; ++j) {
      float* cp = C + (size_t)mbase * ldc + (n0 + 16 * j);
#pragma unroll
      for (int v = 0; v < 8; ++v) cp[(size_t)v * ldc] = acc[i][j][v];
    }
  }
}

// ---------------- decoder: adj = sigmoid(emb @ emb^T), K = 64 --------------
// 32x32 macro-tile per wave (2x2 fragments, K fully unrolled -> 8 wmma),
// bound purely by the 268MB adj_hat store at HBM.
__global__ __launch_bounds__(256) void adj_sigmoid_wmma(
    const __bf16* __restrict__ Eb, float* __restrict__ Out) {
  int wave = (int)((blockIdx.x * blockDim.x + threadIdx.x) >> 5);
  const int tilesN = N_NODES / 32;
  if (wave >= tilesN * tilesN) return;
  int lane = threadIdx.x & 31;
  int tm = wave / tilesN, tn = wave % tilesN;
  int l15 = lane & 15, ksel = lane >> 4;
  int m0 = tm * 32 + l15;
  int n0 = tn * 32 + l15;

  v8f acc[2][2] = {};
#pragma unroll
  for (int k0 = 0; k0 < ZD; k0 += 32) {
    v16bf a[2], b[2];
#pragma unroll
    for (int i = 0; i < 2; ++i)
      a[i] = load_a_frag(Eb, ZD, m0 + 16 * i, k0, ksel);
#pragma unroll
    for (int j = 0; j < 2; ++j)
      b[j] = load_b_frag(Eb, ZD, n0 + 16 * j, k0, ksel);
#pragma unroll
    for (int i = 0; i < 2; ++i)
#pragma unroll
      for (int j = 0; j < 2; ++j)
        acc[i][j] = __builtin_amdgcn_wmma_f32_16x16x32_bf16(
            false, a[i], false, b[j], (short)0, acc[i][j], false, false);
  }
#pragma unroll
  for (int i = 0; i < 2; ++i) {
    int mbase = tm * 32 + 16 * i + 8 * ksel;
#pragma unroll
    for (int j = 0; j < 2; ++j) {
      float* cp = Out + (size_t)mbase * N_NODES + (n0 + 16 * j);
#pragma unroll
      for (int v = 0; v < 8; ++v)
        cp[(size_t)v * N_NODES] = 1.0f / (1.0f + __expf(-acc[i][j][v]));
    }
  }
}

// ---------------- host-side orchestration ----------------
static inline int blocks_for(long long total) {
  return (int)((total + 255) / 256);
}

extern "C" void kernel_launch(void* const* d_in, const int* in_sizes, int n_in,
                              void* d_out, int out_size, void* d_ws, size_t ws_size,
                              hipStream_t stream) {
  (void)in_sizes; (void)n_in; (void)out_size; (void)ws_size;
  const float* x  = (const float*)d_in[0];
  const int* row  = (const int*)d_in[1];
  const int* col  = (const int*)d_in[2];
  const float* ew = (const float*)d_in[3];
  const float* W1 = (const float*)d_in[4];  const float* b1 = (const float*)d_in[5];
  const float* W2 = (const float*)d_in[6];  const float* b2 = (const float*)d_in[7];
  const float* W3 = (const float*)d_in[8];  const float* b3 = (const float*)d_in[9];
  const float* W4 = (const float*)d_in[10]; const float* b4 = (const float*)d_in[11];

  // workspace layout (256B aligned blocks)
  char* ws = (char*)d_ws;
  size_t off = 0;
  auto take = [&](size_t bytes) -> void* {
    void* p = ws + off;
    off += (bytes + 255) & ~(size_t)255;
    return p;
  };
  __bf16* xb   = (__bf16*)take((size_t)N_NODES * IN_PAD * 2);   // 49.3 MB
  __bf16* W1t  = (__bf16*)take((size_t)H1D * IN_PAD * 2);       //  3.1 MB
  __bf16* W2t  = (__bf16*)take((size_t)H2D * H1D * 2);
  __bf16* W3t  = (__bf16*)take((size_t)H3D * H2D * 2);
  __bf16* W4t  = (__bf16*)take((size_t)ZD  * H3D * 2);
  float*  sup  = (float*)take((size_t)N_NODES * H1D * 4);       // 16 MB (reused)
  float*  agg  = (float*)take((size_t)N_NODES * H1D * 4);       // 16 MB (reused)
  __bf16* hb   = (__bf16*)take((size_t)N_NODES * H1D * 2);      //  8 MB (reused)
  __bf16* embb = (__bf16*)take((size_t)N_NODES * ZD * 2);       //  1 MB

  float* emb_out = (float*)d_out;                          // [N, Z]
  float* adj_out = (float*)d_out + (size_t)N_NODES * ZD;   // [N, N]

  // ---- one-time (per call) conversions ----
  {
    long long t = (long long)N_NODES * IN_PAD;
    conv_x_bf16<<<blocks_for(t), 256, 0, stream>>>(x, xb, t);
  }
  { long long t = (long long)H1D * IN_PAD;
    conv_w_bf16_t<<<blocks_for(t), 256, 0, stream>>>(W1, W1t, IN_DIM, H1D, IN_PAD, t); }
  { long long t = (long long)H2D * H1D;
    conv_w_bf16_t<<<blocks_for(t), 256, 0, stream>>>(W2, W2t, H1D, H2D, H1D, t); }
  { long long t = (long long)H3D * H2D;
    conv_w_bf16_t<<<blocks_for(t), 256, 0, stream>>>(W3, W3t, H2D, H3D, H2D, t); }
  { long long t = (long long)ZD * H3D;
    conv_w_bf16_t<<<blocks_for(t), 256, 0, stream>>>(W4, W4t, H3D, ZD, H3D, t); }

  // ---- generic GCN layer runner ----
  auto run_layer = [&](const __bf16* Ain, int K, const __bf16* Wt, int Fo,
                       const float* bias, bool last) {
    int tilesM = N_NODES / 64, tilesN = Fo / 32;   // 64x32 macro-tiles
    int tiles = tilesM * tilesN;
    int blocks = (tiles * 32 + 255) / 256;
    gemm_bf16_wmma<<<blocks, 256, 0, stream>>>(Ain, K, Wt, K, sup, Fo,
                                               tilesN, tiles, K);
    long long nf = (long long)N_NODES * Fo;
    bias_init<<<blocks_for(nf), 256, 0, stream>>>(agg, bias, Fo, nf);
    int fshift4 = (Fo == 512) ? 7 : (Fo == 256) ? 6 : (Fo == 128) ? 5 : 4;
    long long et = (long long)E_EDGES << fshift4;
    spmm_scatter<<<blocks_for(et), 256, 0, stream>>>(row, col, ew, sup, agg,
                                                     Fo, fshift4, et);
    if (!last) {
      relu_to_bf16<<<blocks_for(nf), 256, 0, stream>>>(agg, hb, nf);
    } else {
      relu_emb<<<blocks_for(nf), 256, 0, stream>>>(agg, emb_out, embb, nf);
    }
  };

  run_layer(xb, IN_PAD, W1t, H1D, b1, false);
  run_layer(hb, H1D,    W2t, H2D, b2, false);
  run_layer(hb, H2D,    W3t, H3D, b3, false);
  run_layer(hb, H3D,    W4t, ZD,  b4, true);

  // ---- decoder: adj_hat = sigmoid(emb @ emb^T) ----
  {
    int tiles = (N_NODES / 32) * (N_NODES / 32);  // 32x32 macro-tiles
    int blocks = (tiles * 32 + 255) / 256;
    adj_sigmoid_wmma<<<blocks, 256, 0, stream>>>(embb, adj_out);
  }
}
